// BiLSTM_CRF_2319282340473
// MI455X (gfx1250) — matrix operand order repaired and checked by reading, not compiled, not run on hardware
//
#include <hip/hip_runtime.h>
#include <math.h>

// ---------------------------------------------------------------------------
// BiLSTM-CRF forward for MI455X (gfx1250, wave32, WMMA bf16 16x16x32)
//
// Pipeline (all on `stream`):
//   1) conv_weights : f32 -> bf16 for W_ih / W_hh (both directions)
//   2) gather_embed : X[b*T+t][e] = bf16(emb[x[b,t]][e])
//   3) proj_gemm    : G[dir][m][0:512] = Xbf @ W_ih^T + (b_ih + b_hh)   (WMMA)
//   4) lstm_scan    : sequential over T; h@W_hh^T via WMMA, W_hh register-
//                     resident per wave, c in registers, h via 4KB LDS
//   5) fc_kernel    : em = [hf|hb] @ fc_w^T + fc_b    (K=9, scalar VALU)
//   6) crf_kernel   : per-batch alpha recursion (K=9) + score; atomicAdd NLL
// ---------------------------------------------------------------------------

typedef __bf16 bf16_t;
typedef bf16_t v16bf __attribute__((ext_vector_type(16)));
typedef float  v8f   __attribute__((ext_vector_type(8)));

union Frag { v16bf v; unsigned int u[8]; };

#define Bsz   64
#define Tsz   512
#define Esz   128
#define Hsz   128
#define G4H   512      // 4*H gate width
#define Ksz   9
#define NTOK  (Bsz * Tsz)   // 32768 tokens

__device__ __forceinline__ unsigned short f2bf(float f) {
    union { float f; unsigned int u; } x; x.f = f;
    unsigned int r = x.u + 0x7FFFu + ((x.u >> 16) & 1u);   // round-nearest-even
    return (unsigned short)(r >> 16);
}
__device__ __forceinline__ float sigmoidf_(float x) { return 1.0f / (1.0f + __expf(-x)); }

// -------------------------------------------------------------- prep kernels
__global__ void zero_out_kernel(float* out) {
    if (blockIdx.x == 0 && threadIdx.x == 0) out[0] = 0.0f;
}

__global__ void conv_weights_kernel(const float* wih_f, const float* whh_f,
                                    const float* wih_b, const float* whh_b,
                                    unsigned short* wih_bf, unsigned short* whh_bf) {
    int tid = blockIdx.x * blockDim.x + threadIdx.x;       // 2 * 512*128
    if (tid >= 2 * G4H * Esz) return;
    int dir = tid >> 16;            // 65536 elements per matrix
    int i   = tid & 0xFFFF;
    wih_bf[tid] = f2bf(dir ? wih_b[i] : wih_f[i]);
    whh_bf[tid] = f2bf(dir ? whh_b[i] : whh_f[i]);
}

__global__ void gather_embed_kernel(const float* emb, const int* x, unsigned short* xbf) {
    int tid = blockIdx.x * blockDim.x + threadIdx.x;       // NTOK*128
    if (tid >= NTOK * Esz) return;
    int m = tid >> 7;
    int e = tid & 127;
    xbf[tid] = f2bf(emb[(size_t)x[m] * Esz + e]);
}

// --------------------------------------------- input projection GEMM (WMMA)
// grid = (2048 token tiles, 2 dirs), block = 256 (8 waves).
// Wave w computes n-tiles [4w .. 4w+3] of the 16x512 output strip.
__global__ void proj_gemm_kernel(const unsigned short* xbf, const unsigned short* wih_bf,
                                 const float* bih_f, const float* bhh_f,
                                 const float* bih_b, const float* bhh_b,
                                 float* G) {
    __shared__ unsigned short sA[16 * Esz];                // 4KB token tile (bf16)
    const int mt  = blockIdx.x;
    const int dir = blockIdx.y;
    const unsigned short* w  = wih_bf + (size_t)dir * G4H * Esz;   // [n][k]
    const float* bi = dir ? bih_b : bih_f;
    const float* bh = dir ? bhh_b : bhh_f;
    float* Gd = G + (size_t)dir * NTOK * G4H;

    // stage 16x128 bf16 A tile
    const unsigned int* src = (const unsigned int*)(xbf + (size_t)mt * 16 * Esz);
    unsigned int* dst = (unsigned int*)sA;
    for (int i = threadIdx.x; i < 16 * Esz / 2; i += blockDim.x) dst[i] = src[i];
    __syncthreads();

    const int lane = threadIdx.x & 31;
    const int wv   = threadIdx.x >> 5;
    const int half = lane >> 4;
    const int nl   = lane & 15;

    v8f acc[4];
    const v8f vz = {0.f,0.f,0.f,0.f,0.f,0.f,0.f,0.f};
    #pragma unroll
    for (int q = 0; q < 4; ++q) acc[q] = vz;

    #pragma unroll
    for (int kb = 0; kb < 4; ++kb) {                       // K = 128 = 4 x 32
        Frag a;
        #pragma unroll
        for (int v = 0; v < 8; ++v) {                      // A 16x32 bf16 layout
            int k = kb * 32 + (v & 3) * 2 + (v >> 2) * 16 + 8 * half;
            a.u[v] = *(const unsigned int*)(sA + nl * Esz + k);
        }
        #pragma unroll
        for (int j2 = 0; j2 < 4; ++j2) {
            int nt = wv * 4 + j2;
            Frag b;
            #pragma unroll
            for (int v = 0; v < 8; ++v) {                  // B 32x16 bf16 layout
                int k = kb * 32 + (v & 3) * 2 + (v >> 2) * 16 + 8 * half;
                int n = nt * 16 + nl;
                b.u[v] = *(const unsigned int*)(w + (size_t)n * Esz + k);
            }
            acc[j2] = __builtin_amdgcn_wmma_f32_16x16x32_bf16(
                false, a.v, false, b.v, (short)0, acc[j2], false, false);
        }
    }

    #pragma unroll
    for (int j2 = 0; j2 < 4; ++j2) {
        #pragma unroll
        for (int v = 0; v < 8; ++v) {                      // C/D: m = v+8*half, n = lane&15
            int n = (wv * 4 + j2) * 16 + nl;
            int m = mt * 16 + v + 8 * half;
            Gd[(size_t)m * G4H + n] = acc[j2][v] + bi[n] + bh[n];
        }
    }
}

// ------------------------------------------------------- recurrent LSTM scan
// grid = (4 batch tiles, 2 dirs), block = 256 (8 waves).
// Wave w owns hidden cols j in [16w,16w+16); computes gate tiles at
// n = q*128 + 16w (q = i,f,g,o) so gating is lane-local. W_hh fragments are
// preloaded into registers (16KB/wave); c-state lives in registers; h (bf16)
// round-trips through 4KB LDS each step.
__global__ void lstm_scan_kernel(const unsigned short* whh_bf, const float* G,
                                 float* hsF, float* hsB) {
    __shared__ unsigned short sH[16 * Hsz];                // 4KB
    const int bt  = blockIdx.x;
    const int dir = blockIdx.y;
    const unsigned short* w = whh_bf + (size_t)dir * G4H * Hsz;  // [n][k]
    const float* Gd = G + (size_t)dir * NTOK * G4H;
    float* hs = dir ? hsB : hsF;

    const int lane = threadIdx.x & 31;
    const int wv   = threadIdx.x >> 5;
    const int half = lane >> 4;
    const int nl   = lane & 15;

    // Preload weight-stationary B fragments: 4 gates x 4 k-blocks x 8 dwords
    Frag bfr[4][4];
    #pragma unroll
    for (int q = 0; q < 4; ++q)
        #pragma unroll
        for (int kb = 0; kb < 4; ++kb)
            #pragma unroll
            for (int v = 0; v < 8; ++v) {
                int n = q * 128 + wv * 16 + nl;
                int k = kb * 32 + (v & 3) * 2 + (v >> 2) * 16 + 8 * half;
                bfr[q][kb].u[v] = *(const unsigned int*)(w + (size_t)n * Hsz + k);
            }

    float creg[8];
    #pragma unroll
    for (int v = 0; v < 8; ++v) creg[v] = 0.0f;
    for (int i = threadIdx.x; i < 16 * Hsz; i += blockDim.x) sH[i] = 0;
    __syncthreads();

    const v8f vz = {0.f,0.f,0.f,0.f,0.f,0.f,0.f,0.f};

    for (int s = 0; s < Tsz; ++s) {
        const int tt = dir ? (Tsz - 1 - s) : s;
        v8f acc[4];
        #pragma unroll
        for (int q = 0; q < 4; ++q) acc[q] = vz;

        #pragma unroll
        for (int kb = 0; kb < 4; ++kb) {
            Frag a;
            #pragma unroll
            for (int v = 0; v < 8; ++v) {
                int k = kb * 32 + (v & 3) * 2 + (v >> 2) * 16 + 8 * half;
                a.u[v] = *(const unsigned int*)(sH + nl * Hsz + k);
            }
            #pragma unroll
            for (int q = 0; q < 4; ++q)
                acc[q] = __builtin_amdgcn_wmma_f32_16x16x32_bf16(
                    false, a.v, false, bfr[q][kb].v, (short)0, acc[q], false, false);
        }
        __syncthreads();                                   // all waves done reading sH

        #pragma unroll
        for (int v = 0; v < 8; ++v) {
            int m = v + 8 * half;                          // local batch row
            int b = bt * 16 + m;
            int j = wv * 16 + nl;                          // hidden unit
            const float* grow = Gd + ((size_t)b * Tsz + tt) * G4H;
            float iv = acc[0][v] + grow[j];
            float fv = acc[1][v] + grow[128 + j];
            float gv = acc[2][v] + grow[256 + j];
            float ov = acc[3][v] + grow[384 + j];
            float cn = sigmoidf_(fv) * creg[v] + sigmoidf_(iv) * tanhf(gv);
            creg[v] = cn;
            float h = sigmoidf_(ov) * tanhf(cn);
            sH[m * Hsz + j] = f2bf(h);
            hs[((size_t)b * Tsz + tt) * Hsz + j] = h;
        }
        __syncthreads();                                   // sH ready for next step
    }
}

// ------------------------------------------------------------------- FC (K=9)
__global__ void fc_kernel(const float* hf, const float* hb,
                          const float* fcw, const float* fcb, float* em) {
    int idx = blockIdx.x * blockDim.x + threadIdx.x;
    if (idx >= NTOK * Ksz) return;
    int m = idx / Ksz, k = idx % Ksz;
    const float* hfr = hf + (size_t)m * Hsz;
    const float* hbr = hb + (size_t)m * Hsz;
    const float* wr  = fcw + (size_t)k * (2 * Hsz);
    float s = fcb[k];
    #pragma unroll 4
    for (int j = 0; j < Hsz; ++j) s += hfr[j] * wr[j] + hbr[j] * wr[Hsz + j];
    em[idx] = s;
}

// --------------------------------------------------------------- CRF NLL scan
// One 32-thread block per batch row; threads 0..8 carry alpha[j].
__global__ void crf_kernel(const float* em, const int* y, const unsigned char* mask,
                           const float* start_t, const float* end_t, const float* trans,
                           float* out) {
    __shared__ float alpha[16];
    const int b = blockIdx.x;
    const int j = threadIdx.x;
    const float* e  = em + (size_t)b * Tsz * Ksz;
    const int*   yb = y  + (size_t)b * Tsz;
    const unsigned char* mb = mask + (size_t)b * Tsz;

    if (j < Ksz) alpha[j] = start_t[j] + e[j];
    __syncthreads();

    for (int t = 1; t < Tsz; ++t) {
        float nv = 0.0f;
        if (j < Ksz) {
            float mx = -INFINITY;
            #pragma unroll
            for (int i = 0; i < Ksz; ++i) mx = fmaxf(mx, alpha[i] + trans[i * Ksz + j]);
            float s = 0.0f;
            #pragma unroll
            for (int i = 0; i < Ksz; ++i) s += __expf(alpha[i] + trans[i * Ksz + j] - mx);
            nv = mx + __logf(s) + e[t * Ksz + j];
        }
        __syncthreads();
        if (j < Ksz && mb[t]) alpha[j] = nv;
        __syncthreads();
    }

    if (j == 0) {
        // gold-path score
        int y0 = yb[0];
        float score = start_t[y0] + e[y0];
        int seqlen = 0;
        for (int t = 0; t < Tsz; ++t) seqlen += (int)mb[t];
        int prev = y0;
        for (int t = 1; t < Tsz; ++t) {
            int yt = yb[t];
            float mf = mb[t] ? 1.0f : 0.0f;
            score += (trans[prev * Ksz + yt] + e[t * Ksz + yt]) * mf;
            prev = yt;
        }
        score += end_t[yb[seqlen - 1]];
        // logZ
        float mx = -INFINITY;
        for (int i = 0; i < Ksz; ++i) mx = fmaxf(mx, alpha[i] + end_t[i]);
        float s = 0.0f;
        for (int i = 0; i < Ksz; ++i) s += __expf(alpha[i] + end_t[i] - mx);
        float logZ = mx + __logf(s);
        atomicAdd(out, -(score - logZ));
    }
}

// ------------------------------------------------------------------- launcher
extern "C" void kernel_launch(void* const* d_in, const int* in_sizes, int n_in,
                              void* d_out, int out_size, void* d_ws, size_t ws_size,
                              hipStream_t stream) {
    const float* emb    = (const float*)d_in[0];
    const float* w_ih_f = (const float*)d_in[1];
    const float* w_hh_f = (const float*)d_in[2];
    const float* b_ih_f = (const float*)d_in[3];
    const float* b_hh_f = (const float*)d_in[4];
    const float* w_ih_b = (const float*)d_in[5];
    const float* w_hh_b = (const float*)d_in[6];
    const float* b_ih_b = (const float*)d_in[7];
    const float* b_hh_b = (const float*)d_in[8];
    const float* fc_w   = (const float*)d_in[9];
    const float* fc_b   = (const float*)d_in[10];
    const float* start_t= (const float*)d_in[11];
    const float* end_t  = (const float*)d_in[12];
    const float* trans  = (const float*)d_in[13];
    const int*   x      = (const int*)d_in[14];
    const int*   y      = (const int*)d_in[15];
    const unsigned char* mask = (const unsigned char*)d_in[16];
    float* out = (float*)d_out;

    // workspace layout (bytes)
    char* ws = (char*)d_ws;
    constexpr size_t OFF_XBF = 0;                                    // 32768*128 bf16
    constexpr size_t OFF_WIH = OFF_XBF + (size_t)NTOK * Esz * 2;     // 2*512*128 bf16
    constexpr size_t OFF_WHH = OFF_WIH + (size_t)2 * G4H * Esz * 2;  // 2*512*128 bf16
    constexpr size_t OFF_G   = OFF_WHH + (size_t)2 * G4H * Hsz * 2;  // 2*32768*512 f32
    constexpr size_t OFF_HF  = OFF_G   + (size_t)2 * NTOK * G4H * 4; // 64*512*128 f32
    constexpr size_t OFF_HB  = OFF_HF  + (size_t)NTOK * Hsz * 4;
    constexpr size_t OFF_EM  = OFF_HB  + (size_t)NTOK * Hsz * 4;     // 32768*9 f32

    unsigned short* xbf    = (unsigned short*)(ws + OFF_XBF);
    unsigned short* wih_bf = (unsigned short*)(ws + OFF_WIH);
    unsigned short* whh_bf = (unsigned short*)(ws + OFF_WHH);
    float* G   = (float*)(ws + OFF_G);
    float* hsF = (float*)(ws + OFF_HF);
    float* hsB = (float*)(ws + OFF_HB);
    float* em  = (float*)(ws + OFF_EM);

    zero_out_kernel<<<1, 32, 0, stream>>>(out);

    conv_weights_kernel<<<(2 * G4H * Esz + 255) / 256, 256, 0, stream>>>(
        w_ih_f, w_hh_f, w_ih_b, w_hh_b, wih_bf, whh_bf);

    gather_embed_kernel<<<(NTOK * Esz + 255) / 256, 256, 0, stream>>>(emb, x, xbf);

    proj_gemm_kernel<<<dim3(NTOK / 16, 2), 256, 0, stream>>>(
        xbf, wih_bf, b_ih_f, b_hh_f, b_ih_b, b_hh_b, G);

    lstm_scan_kernel<<<dim3(Bsz / 16, 2), 256, 0, stream>>>(whh_bf, G, hsF, hsB);

    fc_kernel<<<(NTOK * Ksz + 255) / 256, 256, 0, stream>>>(hsF, hsB, fc_w, fc_b, em);

    crf_kernel<<<Bsz, 32, 0, stream>>>(em, y, mask, start_t, end_t, trans, out);
}
